// CrossAttention_5686536699981
// MI455X (gfx1250) — matrix operand order repaired and checked by reading, not compiled, not run on hardware
//
#include <hip/hip_runtime.h>
#include <hip/hip_bf16.h>

typedef __bf16 bf16;
typedef __attribute__((ext_vector_type(16))) __bf16 v16bf;
typedef __attribute__((ext_vector_type(8)))  float  v8f;

#define D_     128
#define HD_    128
#define HEADS_ 4
#define DH_    32
#define NN_    6
#define B_     2
#define Q_     2500
#define KK_    1344
#define NK_    (NN_*KK_)      // 8064
#define EPS_   1e-5f
#define SCALE_ 0.08838834764831845f   // 128^-0.5

static __device__ __forceinline__ int imin(int a, int b) { return a < b ? a : b; }

__device__ __forceinline__ v8f wmma_bf16(v16bf a, v16bf b, v8f c) {
  return __builtin_amdgcn_wmma_f32_16x16x32_bf16(false, a, false, b, (short)0, c, false, false);
}

// A fragment (16x32 bf16) from row-major LDS tile; base -> element (row0, k0), rs = row stride
__device__ __forceinline__ v16bf load_A_tile(const bf16* base, int rs) {
  int ln = threadIdx.x & 31;
  int mr = ln & 15, hf = ln >> 4;
  const bf16* p = base + mr * rs + hf * 16;
  v16bf a;
#pragma unroll
  for (int i = 0; i < 8; ++i) { a[i] = p[i]; a[8 + i] = p[8 + i]; }
  return a;
}

// B fragment from pre-swizzled bf16 weights: one contiguous 32B read per lane.
// Layout: Wz[((ct*KS + ks)*32 + lane)*16 + i]
__device__ __forceinline__ v16bf load_B_swz(const bf16* __restrict__ Wz, int KS, int ks, int ct) {
  int ln = threadIdx.x & 31;
  const v16bf* p = (const v16bf*)(Wz + (((size_t)ct * KS + ks) * 32 + ln) * 16);
  return *p;
}

__device__ __forceinline__ float redmax16(float v) {
#pragma unroll
  for (int m = 1; m < 16; m <<= 1) v = fmaxf(v, __shfl_xor(v, m, 32));
  return v;
}
__device__ __forceinline__ float redsum16(float v) {
#pragma unroll
  for (int m = 1; m < 16; m <<= 1) v += __shfl_xor(v, m, 32);
  return v;
}

// ---------------------------------------------------------------------------
// Kernel 0: convert f32 weight matrix (Kdim x Ncols, row-major) into the
// B-fragment-major bf16 layout: dst[((ct*KS+ks)*32 + lane)*16 + i] where the
// fragment element is W[ks*32 + (lane>>4)*16 + i][ct*16 + (lane&15)].
// ---------------------------------------------------------------------------
__global__ void wswz_kernel(const float* __restrict__ W, bf16* __restrict__ dst,
                            int Kdim, int Ncols) {
  int idx = blockIdx.x * 256 + threadIdx.x;
  if (idx >= Kdim * Ncols) return;
  int KS = Kdim >> 5;
  int i    = idx & 15;
  int ln   = (idx >> 4) & 31;
  int tile = idx >> 9;
  int ks = tile % KS, ct = tile / KS;
  int hf = ln >> 4;
  int k = ks * 32 + hf * 16 + i;
  int j = ct * 16 + (ln & 15);
  dst[idx] = (bf16)W[(size_t)k * Ncols + j];
}

// ---------------------------------------------------------------------------
// Kernel 1: LayerNorm over D + linear (D -> HD) from channels-first input.
// x: (B*N, D, Sp) f32.  mode 0: out (B*N, Sp, HD) bf16 (q,k)
//                        mode 1: out (B, H, Dh, NK) bf16 (v, transposed for attn)
// Input slab is staged into LDS with CDNA5 async global->LDS loads (ASYNCcnt).
// ---------------------------------------------------------------------------
__global__ void proj_ln_kernel(const float* __restrict__ x,
                               const float* __restrict__ lnw, const float* __restrict__ lnb,
                               const bf16* __restrict__ Wz,  const float* __restrict__ bias,
                               bf16* __restrict__ out, int Sp, int mode) {
  __shared__ float xs[16][D_];
  __shared__ bf16  abf[16][D_];
  __shared__ float red_s[16][16];
  __shared__ float red_q[16][16];
  __shared__ float mu_s[16], rs_s[16];

  const int t  = threadIdx.x;
  const int s0 = blockIdx.x * 16;
  const int bn = blockIdx.y;

  // async global -> LDS staging of 16 x 128 f32 (rows clamped at the tail)
#pragma unroll
  for (int i = 0; i < 8; ++i) {
    int flat = t + i * 256;
    int row  = flat & 15;
    int d    = flat >> 4;
    int s    = imin(s0 + row, Sp - 1);
    const float* gp = &x[((size_t)bn * D_ + d) * Sp + s];
    __attribute__((address_space(3))) float* lp =
        (__attribute__((address_space(3))) float*)&xs[row][d];
    asm volatile("global_load_async_to_lds_b32 %0, %1, off"
                 :: "v"(lp), "v"(gp) : "memory");
  }
  asm volatile("s_wait_asynccnt 0" ::: "memory");
  __syncthreads();

  // LayerNorm reductions: 16 segments of 8 per row
  const int row = t & 15, seg = t >> 4;
  {
    float s = 0.f, q = 0.f;
#pragma unroll
    for (int j = 0; j < 8; ++j) { float v = xs[row][seg * 8 + j]; s += v; q += v * v; }
    red_s[row][seg] = s; red_q[row][seg] = q;
  }
  __syncthreads();
  if (seg == 0) {
    float S = 0.f, Qs = 0.f;
#pragma unroll
    for (int i = 0; i < 16; ++i) { S += red_s[row][i]; Qs += red_q[row][i]; }
    float mu  = S * (1.f / D_);
    float var = Qs * (1.f / D_) - mu * mu;
    mu_s[row] = mu;
    rs_s[row] = rsqrtf(var + EPS_);
  }
  __syncthreads();
#pragma unroll
  for (int j = 0; j < 8; ++j) {
    int d = seg * 8 + j;
    float v = (xs[row][d] - mu_s[row]) * rs_s[row] * lnw[d] + lnb[d];
    abf[row][d] = (bf16)v;
  }
  __syncthreads();

  // GEMM: each of 8 waves owns a 16-column tile of HD
  const int wv = t >> 5;
  const int ln = t & 31;
  v8f c = {0.f, 0.f, 0.f, 0.f, 0.f, 0.f, 0.f, 0.f};
#pragma unroll
  for (int ks = 0; ks < 4; ++ks) {
    v16bf a  = load_A_tile(&abf[0][ks * 32], D_);
    v16bf bb = load_B_swz(Wz, 4, ks, wv);
    c = wmma_bf16(a, bb, c);
  }
  const int j  = wv * 16 + (ln & 15);
  const int hf = ln >> 4;
  const float bj = bias[j];
#pragma unroll
  for (int r = 0; r < 8; ++r) {
    int srow = s0 + r + hf * 8;
    if (srow < Sp) {
      float val = c[r] + bj;
      if (mode == 0) {
        out[((size_t)bn * Sp + srow) * HD_ + j] = (bf16)val;
      } else {
        int b = bn / NN_, n = bn % NN_;
        int h = j >> 5, d = j & 31;
        out[(((size_t)b * HEADS_ + h) * DH_ + d) * NK_ + (size_t)n * KK_ + srow] = (bf16)val;
      }
    }
  }
}

// ---------------------------------------------------------------------------
// Kernel 2: flash attention.  One wave = 16 queries x 1 head x 1 batch.
// qh: (B,N,Q,HD) bf16; kh: (B,NK,HD) bf16; vt: (B,H,Dh,NK) bf16; aout: (B,Q,HD) f32
// ---------------------------------------------------------------------------
__global__ void attn_kernel(const bf16* __restrict__ qh, const bf16* __restrict__ khb,
                            const bf16* __restrict__ vtb, float* __restrict__ aout) {
  __shared__ bf16 p_lds[16][32];

  const int ln  = threadIdx.x & 31;
  const int col = ln & 15;
  const int hf  = ln >> 4;
  const int qt  = blockIdx.x;
  const int h   = blockIdx.y;
  const int b   = blockIdx.z;
  const int qg  = imin(qt * 16 + col, Q_ - 1);   // A-fragment row = ln&15

  float mrow[8], lrow[8];
  v8f o0 = {0.f, 0.f, 0.f, 0.f, 0.f, 0.f, 0.f, 0.f};
  v8f o1 = {0.f, 0.f, 0.f, 0.f, 0.f, 0.f, 0.f, 0.f};
#pragma unroll
  for (int r = 0; r < 8; ++r) { mrow[r] = -3.0e38f; lrow[r] = 0.f; }
  const v8f zero = {0.f, 0.f, 0.f, 0.f, 0.f, 0.f, 0.f, 0.f};

  for (int n = 0; n < NN_; ++n) {
    // A fragment: queries for this n-slice (Dh contiguous in memory)
    const bf16* pq = qh + (((size_t)(b * NN_ + n)) * Q_ + qg) * HD_ + h * DH_ + hf * 16;
    v16bf aq;
#pragma unroll
    for (int i = 0; i < 8; ++i) { aq[i] = pq[i]; aq[8 + i] = pq[8 + i]; }

    for (int chunk = 0; chunk < KK_ / 32; ++chunk) {
      const int kgbase = n * KK_ + chunk * 32;

      // K-matrix B fragments (two 16-key tiles), contiguous along Dh
      const bf16* pk0 = khb + ((size_t)b * NK_ + kgbase + col) * HD_ + h * DH_ + hf * 16;
      const bf16* pk1 = pk0 + 16 * HD_;
      v16bf bk0, bk1;
#pragma unroll
      for (int i = 0; i < 16; ++i) { bk0[i] = pk0[i]; bk1[i] = pk1[i]; }

      // prefetch next chunk's K/V while this chunk computes
      if (chunk + 1 < KK_ / 32) {
        __builtin_prefetch(pk0 + 32 * HD_, 0, 3);
        __builtin_prefetch(vtb + (((size_t)b * HEADS_ + h) * DH_ + col) * NK_ +
                               kgbase + 32 + hf * 16, 0, 3);
      }

      v8f s0 = wmma_bf16(aq, bk0, zero);
      v8f s1 = wmma_bf16(aq, bk1, zero);

      // flash-softmax update per row (rows r+hf*8 live in the lane's half)
#pragma unroll
      for (int r = 0; r < 8; ++r) {
        float a0 = s0[r] * SCALE_, a1 = s1[r] * SCALE_;
        float rmax  = redmax16(fmaxf(a0, a1));
        float mnew  = fmaxf(mrow[r], rmax);
        float alpha = __expf(mrow[r] - mnew);
        float p0 = __expf(a0 - mnew);
        float p1 = __expf(a1 - mnew);
        float rsum = redsum16(p0 + p1);
        lrow[r] = lrow[r] * alpha + rsum;
        mrow[r] = mnew;
        o0[r] *= alpha;
        o1[r] *= alpha;
        p_lds[r + hf * 8][col]      = (bf16)p0;
        p_lds[r + hf * 8][16 + col] = (bf16)p1;
      }
      __syncthreads();
      v16bf ap = load_A_tile(&p_lds[0][0], 32);
      __syncthreads();

      // V B fragments: contiguous along keys thanks to (B,H,Dh,NK) layout
      const bf16* pv0 = vtb + (((size_t)b * HEADS_ + h) * DH_ + col)      * NK_ + kgbase + hf * 16;
      const bf16* pv1 = vtb + (((size_t)b * HEADS_ + h) * DH_ + 16 + col) * NK_ + kgbase + hf * 16;
      v16bf bv0, bv1;
#pragma unroll
      for (int i = 0; i < 16; ++i) { bv0[i] = pv0[i]; bv1[i] = pv1[i]; }

      o0 = wmma_bf16(ap, bv0, o0);
      o1 = wmma_bf16(ap, bv1, o1);
    }
  }

#pragma unroll
  for (int r = 0; r < 8; ++r) {
    int srow = qt * 16 + r + hf * 8;
    if (srow < Q_) {
      float inv = 1.f / lrow[r];
      aout[((size_t)b * Q_ + srow) * HD_ + h * DH_ + col]      = o0[r] * inv;
      aout[((size_t)b * Q_ + srow) * HD_ + h * DH_ + 16 + col] = o1[r] * inv;
    }
  }
}

// ---------------------------------------------------------------------------
// Kernel 3: out-proj + skip + LN_pre + MLP(gelu) + residual + LN_post,
// transposed store to (B, D, bH, bW).
// ---------------------------------------------------------------------------
__global__ void tail_kernel(const float* __restrict__ aout, const float* __restrict__ skip,
                            const bf16* __restrict__ wos, const float* __restrict__ bo,
                            const float* __restrict__ lnpw, const float* __restrict__ lnpb,
                            const bf16* __restrict__ w1s, const float* __restrict__ b1,
                            const bf16* __restrict__ w2s, const float* __restrict__ b2,
                            const float* __restrict__ lnow, const float* __restrict__ lnob,
                            float* __restrict__ out) {
  __shared__ bf16  abf[16][D_];
  __shared__ float zf[16][D_];
  __shared__ bf16  zlbf[16][D_];
  __shared__ bf16  hbf[16][2 * D_];
  __shared__ float red_s[16][16];
  __shared__ float red_q[16][16];
  __shared__ float mu_s[16], rs_s[16];

  const int t  = threadIdx.x;
  const int s0 = blockIdx.x * 16;
  const int b  = blockIdx.y;
  const int wv = t >> 5;
  const int ln = t & 31;
  const int hf = ln >> 4;
  const int row = t & 15, seg = t >> 4;

  // load attention output tile, convert to bf16
#pragma unroll
  for (int i = 0; i < 8; ++i) {
    int flat = t + i * 256;
    int rr = flat & 15;
    int d  = flat >> 4;
    int s  = imin(s0 + rr, Q_ - 1);
    abf[rr][d] = (bf16)aout[((size_t)b * Q_ + s) * HD_ + d];
  }
  __syncthreads();

  // GEMM 1: a @ wo + bo + skip
  {
    v8f c = {0.f, 0.f, 0.f, 0.f, 0.f, 0.f, 0.f, 0.f};
#pragma unroll
    for (int ks = 0; ks < 4; ++ks) {
      v16bf a  = load_A_tile(&abf[0][ks * 32], D_);
      v16bf bb = load_B_swz(wos, 4, ks, wv);
      c = wmma_bf16(a, bb, c);
    }
    int j = wv * 16 + (ln & 15);
    float bj = bo[j];
#pragma unroll
    for (int r = 0; r < 8; ++r) {
      int rr = r + hf * 8;
      int s  = imin(s0 + rr, Q_ - 1);
      zf[rr][j] = c[r] + bj + skip[((size_t)b * D_ + j) * Q_ + s];
    }
  }
  __syncthreads();

  // LN_pre
  {
    float s = 0.f, q = 0.f;
#pragma unroll
    for (int j = 0; j < 8; ++j) { float v = zf[row][seg * 8 + j]; s += v; q += v * v; }
    red_s[row][seg] = s; red_q[row][seg] = q;
  }
  __syncthreads();
  if (seg == 0) {
    float S = 0.f, Qs = 0.f;
#pragma unroll
    for (int i = 0; i < 16; ++i) { S += red_s[row][i]; Qs += red_q[row][i]; }
    float mu = S * (1.f / D_);
    float var = Qs * (1.f / D_) - mu * mu;
    mu_s[row] = mu; rs_s[row] = rsqrtf(var + EPS_);
  }
  __syncthreads();
#pragma unroll
  for (int j = 0; j < 8; ++j) {
    int d = seg * 8 + j;
    float v = (zf[row][d] - mu_s[row]) * rs_s[row] * lnpw[d] + lnpb[d];
    zf[row][d]   = v;         // f32 residual
    zlbf[row][d] = (bf16)v;   // GEMM input
  }
  __syncthreads();

  // GEMM 2: zl @ w1 + b1, exact gelu -> hbf (16 x 256); each wave does 2 tiles
#pragma unroll
  for (int tt = 0; tt < 2; ++tt) {
    int ct = wv + tt * 8;
    v8f c = {0.f, 0.f, 0.f, 0.f, 0.f, 0.f, 0.f, 0.f};
#pragma unroll
    for (int ks = 0; ks < 4; ++ks) {
      v16bf a  = load_A_tile(&zlbf[0][ks * 32], D_);
      v16bf bb = load_B_swz(w1s, 4, ks, ct);
      c = wmma_bf16(a, bb, c);
    }
    int j = ct * 16 + (ln & 15);
    float bj = b1[j];
#pragma unroll
    for (int r = 0; r < 8; ++r) {
      float x = c[r] + bj;
      float g = 0.5f * x * (1.f + erff(x * 0.70710678118654752f));
      hbf[r + hf * 8][j] = (bf16)g;
    }
  }
  __syncthreads();

  // GEMM 3: h @ w2 + b2 + residual(zl)
  {
    v8f c = {0.f, 0.f, 0.f, 0.f, 0.f, 0.f, 0.f, 0.f};
#pragma unroll
    for (int ks = 0; ks < 8; ++ks) {
      v16bf a  = load_A_tile(&hbf[0][ks * 32], 2 * D_);
      v16bf bb = load_B_swz(w2s, 8, ks, wv);
      c = wmma_bf16(a, bb, c);
    }
    int j = wv * 16 + (ln & 15);
    float bj = b2[j];
#pragma unroll
    for (int r = 0; r < 8; ++r) {
      int rr = r + hf * 8;
      zf[rr][j] = c[r] + bj + zf[rr][j];
    }
  }
  __syncthreads();

  // LN_post + transposed store
  {
    float s = 0.f, q = 0.f;
#pragma unroll
    for (int j = 0; j < 8; ++j) { float v = zf[row][seg * 8 + j]; s += v; q += v * v; }
    red_s[row][seg] = s; red_q[row][seg] = q;
  }
  __syncthreads();
  if (seg == 0) {
    float S = 0.f, Qs = 0.f;
#pragma unroll
    for (int i = 0; i < 16; ++i) { S += red_s[row][i]; Qs += red_q[row][i]; }
    float mu = S * (1.f / D_);
    float var = Qs * (1.f / D_) - mu * mu;
    mu_s[row] = mu; rs_s[row] = rsqrtf(var + EPS_);
  }
  __syncthreads();
  {
    int s = s0 + row;
    if (s < Q_) {
#pragma unroll
      for (int j = 0; j < 8; ++j) {
        int d = seg * 8 + j;
        float v = (zf[row][d] - mu_s[row]) * rs_s[row] * lnow[d] + lnob[d];
        out[((size_t)b * D_ + d) * Q_ + s] = v;
      }
    }
  }
}

// ---------------------------------------------------------------------------
extern "C" void kernel_launch(void* const* d_in, const int* in_sizes, int n_in,
                              void* d_out, int out_size, void* d_ws, size_t ws_size,
                              hipStream_t stream) {
  (void)in_sizes; (void)n_in; (void)out_size; (void)ws_size;

  const float* q    = (const float*)d_in[0];
  const float* k    = (const float*)d_in[1];
  const float* v    = (const float*)d_in[2];
  const float* skip = (const float*)d_in[3];
  const float* ln_q_w = (const float*)d_in[4];
  const float* ln_q_b = (const float*)d_in[5];
  const float* wq = (const float*)d_in[6];
  const float* bq = (const float*)d_in[7];
  const float* ln_k_w = (const float*)d_in[8];
  const float* ln_k_b = (const float*)d_in[9];
  const float* wk = (const float*)d_in[10];
  const float* bk = (const float*)d_in[11];
  const float* ln_v_w = (const float*)d_in[12];
  const float* ln_v_b = (const float*)d_in[13];
  const float* wv = (const float*)d_in[14];
  const float* bv = (const float*)d_in[15];
  const float* wo = (const float*)d_in[16];
  const float* bo = (const float*)d_in[17];
  const float* lnpw = (const float*)d_in[18];
  const float* lnpb = (const float*)d_in[19];
  const float* w1 = (const float*)d_in[20];
  const float* b1 = (const float*)d_in[21];
  const float* w2 = (const float*)d_in[22];
  const float* b2 = (const float*)d_in[23];
  const float* lnow = (const float*)d_in[24];
  const float* lnob = (const float*)d_in[25];

  char* ws = (char*)d_ws;
  // workspace carve-up (bytes), all offsets 32B-aligned
  const size_t QH_B = (size_t)B_ * NN_ * Q_  * HD_ * 2;   // 7,680,000
  const size_t KH_B = (size_t)B_ * NN_ * KK_ * HD_ * 2;   // 4,128,768
  const size_t AO_B = (size_t)B_ * Q_ * HD_ * 4;          // 2,560,000
  const size_t W128_B = (size_t)D_ * HD_ * 2;             // 32,768 (128x128 bf16)
  const size_t W1_B   = (size_t)D_ * 2 * D_ * 2;          // 65,536
  const size_t W2_B   = (size_t)2 * D_ * D_ * 2;          // 65,536

  bf16*  qhb  = (bf16*)(ws);
  bf16*  khb  = (bf16*)(ws + QH_B);
  bf16*  vtb  = (bf16*)(ws + QH_B + KH_B);
  float* aatt = (float*)(ws + QH_B + 2 * KH_B);
  char*  wbase = ws + QH_B + 2 * KH_B + AO_B;
  bf16* wqs = (bf16*)(wbase);
  bf16* wks = (bf16*)(wbase + W128_B);
  bf16* wvs = (bf16*)(wbase + 2 * W128_B);
  bf16* wos = (bf16*)(wbase + 3 * W128_B);
  bf16* w1s = (bf16*)(wbase + 4 * W128_B);
  bf16* w2s = (bf16*)(wbase + 4 * W128_B + W1_B);

  // weight pre-swizzle to bf16 fragment-major layout
  wswz_kernel<<<dim3(64),  256, 0, stream>>>(wq, wqs, D_, HD_);
  wswz_kernel<<<dim3(64),  256, 0, stream>>>(wk, wks, D_, HD_);
  wswz_kernel<<<dim3(64),  256, 0, stream>>>(wv, wvs, D_, HD_);
  wswz_kernel<<<dim3(64),  256, 0, stream>>>(wo, wos, HD_, D_);
  wswz_kernel<<<dim3(128), 256, 0, stream>>>(w1, w1s, D_, 2 * D_);
  wswz_kernel<<<dim3(128), 256, 0, stream>>>(w2, w2s, 2 * D_, D_);

  const int QT = (Q_ + 15) / 16;   // 157
  const int KT = KK_ / 16;         // 84

  proj_ln_kernel<<<dim3(QT, B_ * NN_), 256, 0, stream>>>(q, ln_q_w, ln_q_b, wqs, bq, qhb, Q_,  0);
  proj_ln_kernel<<<dim3(KT, B_ * NN_), 256, 0, stream>>>(k, ln_k_w, ln_k_b, wks, bk, khb, KK_, 0);
  proj_ln_kernel<<<dim3(KT, B_ * NN_), 256, 0, stream>>>(v, ln_v_w, ln_v_b, wvs, bv, vtb, KK_, 1);
  attn_kernel<<<dim3(QT, HEADS_, B_), 32, 0, stream>>>(qhb, khb, vtb, aatt);
  tail_kernel<<<dim3(QT, B_), 256, 0, stream>>>(aatt, skip, wos, bo, lnpw, lnpb,
                                                w1s, b1, w2s, b2, lnow, lnob, (float*)d_out);
}
